// ProposedConv_16200616641086
// MI455X (gfx1250) — matrix operand order repaired and checked by reading, not compiled, not run on hardware
//
#include <hip/hip_runtime.h>
#include <stdint.h>

// ---------------- problem constants ----------------
#define NN   4096   // nodes
#define EE   2048   // hyperedges
#define DD   256    // feature dim
#define NNZ  65536
#define TEMP_INV 0.0625f   // 1/sqrt(256)
#define ALPHA 0.2f
#define NEG_ATT (-1.0e9f)

typedef __attribute__((ext_vector_type(16))) _Float16 v16h;
typedef __attribute__((ext_vector_type(8)))  float    v8f;
typedef __attribute__((ext_vector_type(4)))  unsigned int v4u;
typedef __attribute__((ext_vector_type(8)))  int      v8i_t;
typedef __attribute__((ext_vector_type(4)))  int      v4i_t;

#if defined(__has_builtin)
#if __has_builtin(__builtin_amdgcn_tensor_load_to_lds) && __has_builtin(__builtin_amdgcn_s_wait_tensorcnt)
#define HAVE_TDM 1
#endif
#endif

// =====================================================================
// f16 GEMM, f32 accumulate:  C[M,N] = A[M,K] @ B[K,N]
// block tile 64x64, 4 waves (128 thr), each wave 32x32 = 2x2 WMMA tiles.
// A tile (64 rows x 32 K) is DMA'd by the Tensor Data Mover straight into
// LDS with hardware padding (4 DWORDs every 16 DWORDs -> 40-half row pitch),
// issued by wave 0 and fenced with s_wait_tensorcnt + workgroup barrier.
// B tile is cooperatively loaded and stored transposed so both WMMA
// fragment reads are contiguous ds_load_b128 matching ISA 16x16x32 layouts.
// M,N multiples of 64; K multiple of 32 (all shapes here satisfy this).
// =====================================================================
__global__ __launch_bounds__(128)
void k_gemm_f16(const _Float16* __restrict__ A, const _Float16* __restrict__ B,
                float* __restrict__ C, int M, int N, int K)
{
    __shared__ __align__(16) _Float16 As[64][40];   // 64 rows x 32 K (pitch 40)
    __shared__ __align__(16) _Float16 Bt[64][40];   // 64 cols x 32 K (transposed)

    const int tid  = threadIdx.x;
    const int lane = tid & 31;
    const int wave = tid >> 5;
    const int wm   = (wave >> 1) * 32;     // wave sub-tile row in block tile
    const int wn   = (wave & 1) * 32;      // wave sub-tile col
    const int m0   = blockIdx.y * 64;
    const int n0   = blockIdx.x * 64;
    const int r16  = lane & 15;
    const int sel  = lane >> 4;            // half-wave selector

    v8f acc[2][2] = {};

#ifdef HAVE_TDM
    // ---- loop-invariant D# group 1 ----
    // [17:16] data_size=1 (2 bytes); [20] pad_enable=1;
    // [24:22] pad_interval=3 (16 DWORDs = one 32-half row);
    // [31:25] pad_amount=3 (4 DWORDs = 8 halfs) -> LDS pitch 40 halfs.
    v8i_t g1;
    g1[0] = (int)0x06D10000u;
    g1[1] = (int)(((unsigned)K & 0xFFFFu) << 16);            // tensor_dim0 lo
    g1[2] = (int)((((unsigned)K >> 16) & 0xFFFFu)            // tensor_dim0 hi
                  | (((unsigned)M & 0xFFFFu) << 16));        // tensor_dim1 lo
    g1[3] = (int)((((unsigned)M >> 16) & 0xFFFFu)            // tensor_dim1 hi
                  | (32u << 16));                            // tile_dim0 = 32
    g1[4] = 64;                                              // tile_dim1 = 64
    g1[5] = K;                                               // dim0_stride lo
    g1[6] = 0; g1[7] = 0;
    v4i_t gz; gz[0] = 0; gz[1] = 0; gz[2] = 0; gz[3] = 0;    // groups 2/3 (2D)
    const unsigned ldsA = (unsigned)(unsigned long long)&As[0][0]; // LDS byte offset
#endif

    for (int k0 = 0; k0 < K; k0 += 32) {
#ifdef HAVE_TDM
        // ---- A tile via Tensor Data Mover (wave 0 issues; EXEC ignored) ----
        if (wave == 0) {
            unsigned long long ga =
                (unsigned long long)(const void*)(A + (size_t)m0 * K + k0);
            v4u g0;
            g0[0] = 1u;                                   // count=1, user mode
            g0[1] = ldsA;                                 // lds_addr
            g0[2] = (unsigned)ga;                         // global_addr lo
            g0[3] = (unsigned)((ga >> 32) & 0x1FFFFFFull) // global_addr hi
                    | (2u << 30);                         // type = 2 (image)
#if __clang_major__ >= 23
            v8i_t z8; z8[0]=0; z8[1]=0; z8[2]=0; z8[3]=0; z8[4]=0; z8[5]=0; z8[6]=0; z8[7]=0;
            __builtin_amdgcn_tensor_load_to_lds(g0, g1, gz, gz, z8, 0);
#else
            __builtin_amdgcn_tensor_load_to_lds(g0, g1, gz, gz, 0);
#endif
        }
#else
        // ---- fallback: cooperative A tile load 64x32 (2 x b128 per thread) ----
        #pragma unroll
        for (int i = 0; i < 2; ++i) {
            int c   = tid + i * 128;       // chunk 0..255
            int row = c >> 2;
            int col = (c & 3) * 8;
            uint4 v = *(const uint4*)(A + (size_t)(m0 + row) * K + k0 + col);
            *(uint4*)(&As[row][col]) = v;
        }
#endif
        // prefetch next B tile (global_prefetch_b8)
        if (k0 + 32 < K)
            __builtin_prefetch(B + (size_t)(k0 + 32 + (tid >> 3)) * N + n0, 0, 0);

        // ---- load B tile 32x64, store transposed ----
        #pragma unroll
        for (int i = 0; i < 2; ++i) {
            int c   = tid + i * 128;
            int kk  = c >> 3;
            int col = (c & 7) * 8;
            union { uint4 q; _Float16 h[8]; } u;
            u.q = *(const uint4*)(B + (size_t)(k0 + kk) * N + n0 + col);
            #pragma unroll
            for (int j = 0; j < 8; ++j) Bt[col + j][kk] = u.h[j];
        }
#ifdef HAVE_TDM
        if (wave == 0) __builtin_amdgcn_s_wait_tensorcnt(0);
#endif
        __syncthreads();

        // ---- fragments ----
        // A lane layout: row = lane&15 ; K = sel*8 + {0..7} and sel*8+16+{0..7}
        v16h aF[2], bF[2];
        #pragma unroll
        for (int i = 0; i < 2; ++i) {
            union { v16h v; uint4 q[2]; } ua;
            const _Float16* ap = &As[wm + i * 16 + r16][sel * 8];
            ua.q[0] = *(const uint4*)(ap);
            ua.q[1] = *(const uint4*)(ap + 16);
            aF[i] = ua.v;
        }
        // B lane layout: col = lane&15 ; K = sel*8 + {0..7} and sel*8+16+{0..7}
        #pragma unroll
        for (int j = 0; j < 2; ++j) {
            union { v16h v; uint4 q[2]; } ub;
            const _Float16* bp = &Bt[wn + j * 16 + r16][sel * 8];
            ub.q[0] = *(const uint4*)(bp);
            ub.q[1] = *(const uint4*)(bp + 16);
            bF[j] = ub.v;
        }
        #pragma unroll
        for (int i = 0; i < 2; ++i)
            #pragma unroll
            for (int j = 0; j < 2; ++j)
                acc[i][j] = __builtin_amdgcn_wmma_f32_16x16x32_f16(
                    false, aF[i], false, bF[j], (short)0, acc[i][j], false, false);
        __syncthreads();
    }

    // ---- epilogue: C/D layout: N=lane&15, M = sel*8 + vgpr ----
    const int mb = sel * 8;
    #pragma unroll
    for (int i = 0; i < 2; ++i)
        #pragma unroll
        for (int j = 0; j < 2; ++j) {
            int cm = m0 + wm + i * 16 + mb;
            int cn = n0 + wn + j * 16 + r16;
            float* cp = C + (size_t)cm * N + cn;
            #pragma unroll
            for (int v = 0; v < 8; ++v)
                cp[(size_t)v * N] = acc[i][j][v];
        }
}

// ---------------- scatter incidence: H [N,E] f16, Ht [E,N] f16, bitmasks ----------------
__global__ void k_scatter(const int* __restrict__ nidx, const int* __restrict__ eidx,
                          _Float16* __restrict__ H, _Float16* __restrict__ Ht,
                          unsigned* __restrict__ bm)
{
    int t = blockIdx.x * blockDim.x + threadIdx.x;
    if (t >= NNZ) return;
    int n = nidx[t], e = eidx[t];
    H[(size_t)n * EE + e]  = (_Float16)1.0f;
    Ht[(size_t)e * NN + n] = (_Float16)1.0f;
    atomicOr(&bm[n * 64 + (e >> 5)], 1u << (e & 31));
}

// ---------------- f32 -> f16 convert (optional scale) ----------------
__global__ void k_cvt(const float* __restrict__ s, _Float16* __restrict__ d,
                      float scale, int n)
{
    int t = blockIdx.x * blockDim.x + threadIdx.x;
    if (t < n) d[t] = (_Float16)(s[t] * scale);
}

// ---------------- xw + bias -> f16 ----------------
__global__ void k_bias_cvt(const float* __restrict__ s, const float* __restrict__ bias,
                           _Float16* __restrict__ d, int n)
{
    int t = blockIdx.x * blockDim.x + threadIdx.x;
    if (t < n) d[t] = (_Float16)(s[t] + bias[t & (DD - 1)]);
}

// ---------------- tiled transpose: dst[C,R] = src[R,C]^T ----------------
template <typename TIN, typename TOUT>
__global__ void k_transpose(const TIN* __restrict__ src, TOUT* __restrict__ dst,
                            int R, int C)
{
    __shared__ TIN tile[32][33];
    int c0 = blockIdx.x * 32, r0 = blockIdx.y * 32;
    int tx = threadIdx.x, ty = threadIdx.y;      // (32,8)
    #pragma unroll
    for (int i = 0; i < 32; i += 8)
        tile[ty + i][tx] = src[(size_t)(r0 + ty + i) * C + c0 + tx];
    __syncthreads();
    #pragma unroll
    for (int i = 0; i < 32; i += 8)
        dst[(size_t)(c0 + ty + i) * R + r0 + tx] = (TOUT)tile[tx][ty + i];
}

// ---------------- per-node dots: Wh1, Wh2, s1 ----------------
__global__ __launch_bounds__(256)
void k_dots(const float* __restrict__ WhG, const float* __restrict__ x4,
            const float* __restrict__ ag, const float* __restrict__ wc,
            float* __restrict__ Wh1, float* __restrict__ Wh2, float* __restrict__ s1)
{
    __shared__ float r1[256], r2[256], r3[256];
    int n = blockIdx.x, t = threadIdx.x;
    float a = WhG[(size_t)n * DD + t];
    r1[t] = a * ag[t];
    r2[t] = a * ag[DD + t];
    r3[t] = x4[(size_t)n * DD + t] * wc[t];
    __syncthreads();
    for (int o = 128; o > 0; o >>= 1) {
        if (t < o) { r1[t] += r1[t + o]; r2[t] += r2[t + o]; r3[t] += r3[t + o]; }
        __syncthreads();
    }
    if (t == 0) { Wh1[n] = r1[0]; Wh2[n] = r2[0]; s1[n] = r3[0] * TEMP_INV; }
}

// ---------------- attention_g row softmax with bitmask adjacency ----------------
__global__ __launch_bounds__(256)
void k_attng(const float* __restrict__ Wh1, const float* __restrict__ Wh2,
             const unsigned* __restrict__ bm, _Float16* __restrict__ attng)
{
    __shared__ __align__(16) unsigned bmI[64];
    __shared__ float red[256];
    int i = blockIdx.x, t = threadIdx.x;
    if (t < 64) bmI[t] = bm[i * 64 + t];
    __syncthreads();

    float w1 = Wh1[i];
    float e[16];
    float m = -3.0e38f;
    #pragma unroll
    for (int s = 0; s < 16; ++s) {
        int j = s * 256 + t;
        unsigned any = 0;
        const uint4* bj4 = (const uint4*)(bm + (size_t)j * 64);
        #pragma unroll
        for (int w = 0; w < 16; ++w) {
            uint4 q  = bj4[w];
            uint4 bi = *(const uint4*)(&bmI[w * 4]);
            any |= (q.x & bi.x) | (q.y & bi.y) | (q.z & bi.z) | (q.w & bi.w);
        }
        bool adj = (any != 0u) || (j == i);
        float ev;
        if (adj) { float x = w1 + Wh2[j]; ev = x > 0.f ? x : ALPHA * x; }
        else       ev = -3.0e38f;
        e[s] = ev; m = fmaxf(m, ev);
    }
    red[t] = m; __syncthreads();
    for (int o = 128; o > 0; o >>= 1) { if (t < o) red[t] = fmaxf(red[t], red[t + o]); __syncthreads(); }
    m = red[0]; __syncthreads();
    float sum = 0.f;
    #pragma unroll
    for (int s = 0; s < 16; ++s) {
        float p = (e[s] <= -1.0e37f) ? 0.f : __expf(e[s] - m);
        e[s] = p; sum += p;
    }
    red[t] = sum; __syncthreads();
    for (int o = 128; o > 0; o >>= 1) { if (t < o) red[t] += red[t + o]; __syncthreads(); }
    float inv = 1.0f / red[0];
    #pragma unroll
    for (int s = 0; s < 16; ++s)
        attng[(size_t)i * NN + s * 256 + t] = (_Float16)(e[s] * inv);
}

// ---------------- att1[e,n] = softmax_n(g2hT) + softmax_n(masked s1) ----------------
__global__ __launch_bounds__(256)
void k_att1(const float* __restrict__ g2hT, const _Float16* __restrict__ Ht,
            const float* __restrict__ s1, _Float16* __restrict__ att1)
{
    __shared__ float red[256];
    int e = blockIdx.x, t = threadIdx.x;
    float g[16], vv[16];
    float m1 = -3.0e38f, m2 = -3.0e38f;
    #pragma unroll
    for (int s = 0; s < 16; ++s) {
        int n = s * 256 + t;
        g[s]  = g2hT[(size_t)e * NN + n];
        float hv = (float)Ht[(size_t)e * NN + n];
        vv[s] = (hv != 0.f) ? s1[n] : NEG_ATT;
        m1 = fmaxf(m1, g[s]); m2 = fmaxf(m2, vv[s]);
    }
    red[t] = m1; __syncthreads();
    for (int o = 128; o > 0; o >>= 1) { if (t < o) red[t] = fmaxf(red[t], red[t + o]); __syncthreads(); }
    m1 = red[0]; __syncthreads();
    red[t] = m2; __syncthreads();
    for (int o = 128; o > 0; o >>= 1) { if (t < o) red[t] = fmaxf(red[t], red[t + o]); __syncthreads(); }
    m2 = red[0]; __syncthreads();
    float sA = 0.f, sB = 0.f;
    #pragma unroll
    for (int s = 0; s < 16; ++s) { g[s] = __expf(g[s] - m1); sA += g[s];
                                   vv[s] = __expf(vv[s] - m2); sB += vv[s]; }
    red[t] = sA; __syncthreads();
    for (int o = 128; o > 0; o >>= 1) { if (t < o) red[t] += red[t + o]; __syncthreads(); }
    sA = red[0]; __syncthreads();
    red[t] = sB; __syncthreads();
    for (int o = 128; o > 0; o >>= 1) { if (t < o) red[t] += red[t + o]; __syncthreads(); }
    sB = red[0];
    float iA = 1.0f / sA, iB = 1.0f / sB;
    #pragma unroll
    for (int s = 0; s < 16; ++s)
        att1[(size_t)e * NN + s * 256 + t] = (_Float16)(g[s] * iA + vv[s] * iB);
}

// ---------------- att2hn[n,e] = softmax_e(g2h) + softmax_e(masked attn2) ----------------
__global__ __launch_bounds__(256)
void k_att2hn(const float* __restrict__ g2h, const float* __restrict__ attn2,
              const _Float16* __restrict__ H, _Float16* __restrict__ out)
{
    __shared__ float red[256];
    int n = blockIdx.x, t = threadIdx.x;
    float g[8], vv[8];
    float m1 = -3.0e38f, m2 = -3.0e38f;
    #pragma unroll
    for (int s = 0; s < 8; ++s) {
        int e = s * 256 + t;
        g[s]  = g2h[(size_t)n * EE + e];
        float hv = (float)H[(size_t)n * EE + e];
        vv[s] = (hv != 0.f) ? attn2[(size_t)n * EE + e] : NEG_ATT;
        m1 = fmaxf(m1, g[s]); m2 = fmaxf(m2, vv[s]);
    }
    red[t] = m1; __syncthreads();
    for (int o = 128; o > 0; o >>= 1) { if (t < o) red[t] = fmaxf(red[t], red[t + o]); __syncthreads(); }
    m1 = red[0]; __syncthreads();
    red[t] = m2; __syncthreads();
    for (int o = 128; o > 0; o >>= 1) { if (t < o) red[t] = fmaxf(red[t], red[t + o]); __syncthreads(); }
    m2 = red[0]; __syncthreads();
    float sA = 0.f, sB = 0.f;
    #pragma unroll
    for (int s = 0; s < 8; ++s) { g[s] = __expf(g[s] - m1); sA += g[s];
                                  vv[s] = __expf(vv[s] - m2); sB += vv[s]; }
    red[t] = sA; __syncthreads();
    for (int o = 128; o > 0; o >>= 1) { if (t < o) red[t] += red[t + o]; __syncthreads(); }
    sA = red[0]; __syncthreads();
    red[t] = sB; __syncthreads();
    for (int o = 128; o > 0; o >>= 1) { if (t < o) red[t] += red[t + o]; __syncthreads(); }
    sB = red[0];
    float iA = 1.0f / sA, iB = 1.0f / sB;
    #pragma unroll
    for (int s = 0; s < 8; ++s)
        out[(size_t)n * EE + s * 256 + t] = (_Float16)(g[s] * iA + vv[s] * iB);
}

// ---------------- final: node=relu, h_prime=elu(gWh+t2b+t3b) ----------------
__global__ void k_final(const float* __restrict__ nodeF, const float* __restrict__ gWh,
                        const float* __restrict__ t2b, const float* __restrict__ t3b,
                        float* __restrict__ out, int n)
{
    int t = blockIdx.x * blockDim.x + threadIdx.x;
    if (t >= n) return;
    float nv = nodeF[t];
    out[t] = nv > 0.f ? nv : 0.f;
    float h = gWh[t] + t2b[t] + t3b[t];
    out[n + t] = h > 0.f ? h : (__expf(h) - 1.0f);
}

// =====================================================================
extern "C" void kernel_launch(void* const* d_in, const int* in_sizes, int n_in,
                              void* d_out, int out_size, void* d_ws, size_t ws_size,
                              hipStream_t stream)
{
    (void)in_sizes; (void)n_in; (void)out_size; (void)ws_size;
    const float* x   = (const float*)d_in[0];
    const float* w   = (const float*)d_in[1];
    const float* w2  = (const float*)d_in[2];
    const float* w3  = (const float*)d_in[3];
    const float* wg  = (const float*)d_in[4];
    const float* ag  = (const float*)d_in[5];
    const float* wc  = (const float*)d_in[6];
    const float* bias= (const float*)d_in[7];
    const int*   hei = (const int*)d_in[8];       // [2, NNZ]: row0 node, row1 edge
    float* out = (float*)d_out;

    // ---- workspace carve-out ----
    char* p = (char*)d_ws;
    auto alloc = [&](size_t bytes) -> void* {
        void* r = (void*)p; p += (bytes + 255) & ~(size_t)255; return r;
    };
    _Float16* H      = (_Float16*)alloc((size_t)NN * EE * 2);
    _Float16* Ht     = (_Float16*)alloc((size_t)EE * NN * 2);
    unsigned* bm     = (unsigned*)alloc((size_t)NN * 64 * 4);
    _Float16* xh     = (_Float16*)alloc((size_t)NN * DD * 2);
    _Float16* wgh    = (_Float16*)alloc((size_t)DD * DD * 2);
    _Float16* w2h    = (_Float16*)alloc((size_t)DD * DD * 2);
    _Float16* wh     = (_Float16*)alloc((size_t)DD * DD * 2);
    _Float16* w3h    = (_Float16*)alloc((size_t)DD * DD * 2);
    float*    WhG    = (float*)alloc((size_t)NN * DD * 4);
    _Float16* WhGh   = (_Float16*)alloc((size_t)NN * DD * 2);
    float*    x4     = (float*)alloc((size_t)NN * DD * 4);
    _Float16* x4sh   = (_Float16*)alloc((size_t)NN * DD * 2);
    float*    xwf    = (float*)alloc((size_t)NN * DD * 4);
    _Float16* xwh    = (_Float16*)alloc((size_t)NN * DD * 2);
    float*    Wh1    = (float*)alloc(NN * 4);
    float*    Wh2    = (float*)alloc(NN * 4);
    float*    s1     = (float*)alloc(NN * 4);
    _Float16* attng  = (_Float16*)alloc((size_t)NN * NN * 2);
    float*    g2h    = (float*)alloc((size_t)NN * EE * 4);
    float*    g2hT   = (float*)alloc((size_t)EE * NN * 4);
    float*    gWh    = (float*)alloc((size_t)NN * DD * 4);
    _Float16* att1   = (_Float16*)alloc((size_t)EE * NN * 2);
    _Float16* att1T  = (_Float16*)alloc((size_t)NN * EE * 2);
    float*    edgeF  = (float*)alloc((size_t)EE * DD * 4);
    _Float16* edgeH  = (_Float16*)alloc((size_t)EE * DD * 2);
    float*    e4     = (float*)alloc((size_t)EE * DD * 4);
    _Float16* e4Th   = (_Float16*)alloc((size_t)DD * EE * 2);
    float*    attn2  = (float*)alloc((size_t)NN * EE * 4);
    _Float16* a2hn   = (_Float16*)alloc((size_t)NN * EE * 2);
    _Float16* a2hnT  = (_Float16*)alloc((size_t)EE * NN * 2);
    float*    nodeF  = (float*)alloc((size_t)NN * DD * 4);
    float*    t2a    = (float*)alloc((size_t)EE * DD * 4);
    _Float16* t2aH   = (_Float16*)alloc((size_t)EE * DD * 2);
    float*    t2b    = (float*)alloc((size_t)NN * DD * 4);
    float*    t3a    = (float*)alloc((size_t)EE * DD * 4);
    _Float16* t3aH   = (_Float16*)alloc((size_t)EE * DD * 2);
    float*    t3b    = (float*)alloc((size_t)NN * DD * 4);

    auto gemm = [&](const _Float16* A, const _Float16* B, float* C, int M, int N, int K) {
        dim3 g(N / 64, M / 64);
        k_gemm_f16<<<g, 128, 0, stream>>>(A, B, C, M, N, K);
    };
    auto cvt = [&](const float* s, _Float16* d, float scale, int n) {
        k_cvt<<<(n + 255) / 256, 256, 0, stream>>>(s, d, scale, n);
    };

    // ---- 0: incidence structures ----
    hipMemsetAsync(H,  0, (size_t)NN * EE * 2, stream);
    hipMemsetAsync(Ht, 0, (size_t)EE * NN * 2, stream);
    hipMemsetAsync(bm, 0, (size_t)NN * 64 * 4, stream);
    k_scatter<<<NNZ / 256, 256, 0, stream>>>(hei, hei + NNZ, H, Ht, bm);

    // ---- 1: convert inputs to f16, input projections ----
    cvt(x,  xh,  1.0f, NN * DD);
    cvt(wg, wgh, 1.0f, DD * DD);
    cvt(w2, w2h, 1.0f, DD * DD);
    cvt(w,  wh,  1.0f, DD * DD);
    cvt(w3, w3h, 1.0f, DD * DD);
    gemm(xh, wgh, WhG, NN, DD, DD);      // Wh_g
    gemm(xh, w2h, x4,  NN, DD, DD);      // x_4att
    gemm(xh, wh,  xwf, NN, DD, DD);      // x@weight
    cvt(WhG, WhGh, 1.0f, NN * DD);
    cvt(x4, x4sh, TEMP_INV, NN * DD);    // x_4att / temp
    k_bias_cvt<<<(NN * DD + 255) / 256, 256, 0, stream>>>(xwf, bias, xwh, NN * DD);

    // ---- 2: GAT scores + masked softmax via bitmask adjacency ----
    k_dots<<<NN, 256, 0, stream>>>(WhG, x4, ag, wc, Wh1, Wh2, s1);
    k_attng<<<NN, 256, 0, stream>>>(Wh1, Wh2, bm, attng);

    // ---- 3: the big GEMMs off attention_g ----
    gemm(attng, H,    g2h, NN, EE, NN);  // graph2hyper [N,E]
    gemm(attng, WhGh, gWh, NN, DD, NN);  // attention_g @ Wh_g term
    {
        dim3 g(EE / 32, NN / 32);
        k_transpose<float, float><<<g, dim3(32, 8), 0, stream>>>(g2h, g2hT, NN, EE);
    }

    // ---- 4: att1 / edge branch ----
    k_att1<<<EE, 256, 0, stream>>>(g2hT, Ht, s1, att1);
    {
        dim3 g(NN / 32, EE / 32);
        k_transpose<_Float16, _Float16><<<g, dim3(32, 8), 0, stream>>>(att1, att1T, EE, NN);
    }
    gemm(att1, xwh, edgeF, EE, DD, NN);  // edge = att1 @ xw
    cvt(edgeF, edgeH, 1.0f, EE * DD);
    gemm(edgeH, w3h, e4, EE, DD, DD);    // edge_4att
    {
        dim3 g(DD / 32, EE / 32);
        k_transpose<float, _Float16><<<g, dim3(32, 8), 0, stream>>>(e4, e4Th, EE, DD);
    }
    gemm(x4sh, e4Th, attn2, NN, EE, DD); // (x_4att/temp) @ edge_4att.T

    // ---- 5: att2hn / node branch ----
    k_att2hn<<<NN, 256, 0, stream>>>(g2h, attn2, H, a2hn);
    {
        dim3 g(EE / 32, NN / 32);
        k_transpose<_Float16, _Float16><<<g, dim3(32, 8), 0, stream>>>(a2hn, a2hnT, NN, EE);
    }
    gemm(a2hn, edgeH, nodeF, NN, DD, EE);   // node = att2hn @ edge

    // ---- 6: factored Gram terms: (att1.T att1)Wh = att1.T (att1 Wh), etc ----
    gemm(att1, WhGh, t2a, EE, DD, NN);
    cvt(t2a, t2aH, 1.0f, EE * DD);
    gemm(att1T, t2aH, t2b, NN, DD, EE);
    gemm(a2hnT, WhGh, t3a, EE, DD, NN);
    cvt(t3a, t3aH, 1.0f, EE * DD);
    gemm(a2hn, t3aH, t3b, NN, DD, EE);

    // ---- 7: activations + output ----
    k_final<<<(NN * DD + 255) / 256, 256, 0, stream>>>(nodeF, gWh, t2b, t3b, out, NN * DD);
}
